// DharmaAttention_78632261255399
// MI455X (gfx1250) — matrix operand order repaired and checked by reading, not compiled, not run on hardware
//
#include <hip/hip_runtime.h>
#include <hip/hip_bf16.h>
#include <stdint.h>

typedef unsigned short u16;
typedef unsigned int   u32;
typedef unsigned long long u64;
typedef __attribute__((ext_vector_type(16))) __bf16 v16bf;
typedef __attribute__((ext_vector_type(8)))  float  v8f;

#define BATCH 2
#define SEQ   2048
#define HID   2048
#define NHEAD 16
#define HDIM  128
#define MTOT  (BATCH*SEQ)

__device__ __forceinline__ u16 f2bf(float f) {
  u32 u = __builtin_bit_cast(u32, f);
  u32 r = (u + 0x7FFFu + ((u >> 16) & 1u)) >> 16;
  return (u16)r;
}
__device__ __forceinline__ float bf2f(u16 h) {
  u32 u = ((u32)h) << 16;
  return __builtin_bit_cast(float, u);
}

union FragB { u32 u[8]; v16bf v; };

// ---- CDNA5 async copy global -> LDS (16B per lane), tracked on ASYNCcnt ----
__device__ __forceinline__ u32 lds_byte_off(const void* p) {
  return (u32)(u64)p;                 // LDS aperture: addr[31:0] == LDS offset
}
__device__ __forceinline__ void async_copy_b128(u32 ldsOff, const void* gaddr) {
  asm volatile("global_load_async_to_lds_b128 %0, %1, off"
               :: "v"(ldsOff), "v"((u64)gaddr) : "memory");
}
__device__ __forceinline__ void wait_async0() {
  asm volatile("s_wait_asynccnt 0x0" ::: "memory");
}

// ---------------- elementwise f32 -> bf16 ----------------
__global__ void k_cvt_bf16(const float* __restrict__ in, u16* __restrict__ out, int n) {
  int i = blockIdx.x * blockDim.x + threadIdx.x;
  if (i < n) out[i] = f2bf(in[i]);
}

// ------------- W[o][h] (f32) -> Wt[h][o] (bf16), tiled transpose -------------
__global__ void k_transpose_w(const float* __restrict__ W, u16* __restrict__ Wt) {
  __shared__ float t[32][33];
  int tx = threadIdx.x, ty = threadIdx.y;
  int n0 = blockIdx.y * 32, k0 = blockIdx.x * 32;
#pragma unroll
  for (int j = 0; j < 4; ++j)
    t[ty + j*8][tx] = W[(size_t)(n0 + ty + j*8) * HID + k0 + tx];
  __syncthreads();
#pragma unroll
  for (int j = 0; j < 4; ++j)
    Wt[(size_t)(k0 + ty + j*8) * HID + n0 + tx] = f2bf(t[tx][ty + j*8]);
}

// ------------- bf16 WMMA GEMM: C = A(M x K) * Bt(K x N) -------------
// MODE 0: write bf16 to [B,NH,S,HD] layout (QKV).  MODE 1: write f32 row-major (final).
template <int MODE>
__global__ __launch_bounds__(256) void k_gemm(const u16* __restrict__ A,
                                              const u16* __restrict__ Bt,
                                              u16* __restrict__ outBf,
                                              float* __restrict__ outF) {
  __shared__ u16 lA[128][32];   // [m][k]
  __shared__ u16 lB[128][32];   // [n][k]  (transposed at fill so k-pairs are contiguous)
  int tid  = threadIdx.x;
  int lane = tid & 31, wid = tid >> 5;
  int wm = wid & 3, wn = wid >> 2;       // 4 x 2 wave grid -> 128 x 128 block tile
  int hf = lane >> 4, mrow = lane & 15;
  int m0 = blockIdx.y * 128, n0 = blockIdx.x * 128;

  v8f acc[2][4];
#pragma unroll
  for (int a = 0; a < 2; ++a)
#pragma unroll
    for (int c = 0; c < 4; ++c) acc[a][c] = {};

  for (int k0 = 0; k0 < HID; k0 += 32) {
    // A tile: 128 rows x 32 k -- direct copy via async-to-LDS (no VGPR staging)
#pragma unroll
    for (int it = 0; it < 2; ++it) {
      int slot = it * 256 + tid;
      int row = slot >> 2, cg = slot & 3;
      async_copy_b128(lds_byte_off(&lA[row][cg*8]),
                      &A[(size_t)(m0+row)*HID + k0 + cg*8]);
    }
    // B tile: 32 k x 128 n, VGPR-staged transpose into [n][k]
#pragma unroll
    for (int it = 0; it < 2; ++it) {
      int slot = it * 256 + tid;
      int kr = slot >> 4, ng = slot & 15;
      uint4 q = *(const uint4*)&Bt[(size_t)(k0+kr)*HID + n0 + ng*8];
      const u16* e = (const u16*)&q;
#pragma unroll
      for (int jj = 0; jj < 8; ++jj) lB[ng*8 + jj][kr] = e[jj];
    }
    // prefetch next k-step into GL2 while this one computes
    if (k0 + 32 < HID) {
      __builtin_prefetch(&A[(size_t)(m0 + (tid >> 1)) * HID + k0 + 32], 0, 0);
      __builtin_prefetch(&Bt[(size_t)(k0 + 32 + (tid >> 3)) * HID + n0], 0, 0);
    }
    wait_async0();
    __syncthreads();

    FragB af[2], bfr[4];
#pragma unroll
    for (int mt = 0; mt < 2; ++mt)
#pragma unroll
      for (int v = 0; v < 8; ++v) {
        int k = ((v >> 2) << 4) + hf*8 + ((v & 3) << 1);   // A-layout k index
        af[mt].u[v] = *(const u32*)&lA[wm*32 + mt*16 + mrow][k];
      }
#pragma unroll
    for (int nt = 0; nt < 4; ++nt)
#pragma unroll
      for (int v = 0; v < 8; ++v) {
        int k = ((v >> 2) << 4) + hf*8 + ((v & 3) << 1);   // A-layout k index
        bfr[nt].u[v] = *(const u32*)&lB[wn*64 + nt*16 + mrow][k];
      }
#pragma unroll
    for (int mt = 0; mt < 2; ++mt)
#pragma unroll
      for (int nt = 0; nt < 4; ++nt)
        acc[mt][nt] = __builtin_amdgcn_wmma_f32_16x16x32_bf16(
            false, af[mt].v, false, bfr[nt].v, (short)0, acc[mt][nt], false, false);
    __syncthreads();
  }

#pragma unroll
  for (int mt = 0; mt < 2; ++mt)
#pragma unroll
    for (int nt = 0; nt < 4; ++nt)
#pragma unroll
      for (int r = 0; r < 8; ++r) {
        int m = m0 + wm*32 + mt*16 + hf*8 + r;   // C layout: m = (lane/16)*8 + r
        int n = n0 + wn*64 + nt*16 + mrow;       // n = lane%16
        float val = acc[mt][nt][r];
        if (MODE == 0) {
          int b = m >> 11, s = m & (SEQ - 1);
          int head = n >> 7, d = n & 127;
          outBf[(((size_t)(b*NHEAD + head) * SEQ + s) << 7) + d] = f2bf(val);
        } else {
          outF[(size_t)m * HID + n] = val;
        }
      }
}

// ------------- RoPE in place on bf16 [B,NH,S,HD] -------------
__global__ void k_rope(u16* __restrict__ X) {
  int idx = blockIdx.x * blockDim.x + threadIdx.x;   // B*NH*S*64 threads
  int d  = idx & 63;
  int s  = (idx >> 6) & (SEQ - 1);
  int bh = idx >> 17;
  size_t base = ((size_t)bh * SEQ + s) << 7;
  // inv_freq = theta^(-2d/128) = exp(-2d/128 * ln(10000))
  float inv = __expf(-(float)(2*d) * (9.2103403719761836f / 128.0f));
  float ang = (float)s * inv;
  float sn, cn;
  __sincosf(ang, &sn, &cn);
  float x1 = bf2f(X[base + d]);
  float x2 = bf2f(X[base + d + 64]);
  X[base + d]      = f2bf(x1*cn - x2*sn);
  X[base + d + 64] = f2bf(x2*cn + x1*sn);
}

// ------------- causal flash attention, one (b,h,128 q-rows) per block -------------
__global__ __launch_bounds__(256) void k_flash(const u16* __restrict__ Q,
                                               const u16* __restrict__ K,
                                               const u16* __restrict__ V,
                                               u16* __restrict__ O) {
  __shared__ u16 lK[64][128];      // [key][d]  -> d-pairs contiguous (B-frag for QK^T)
  __shared__ u16 lVt[128][64];     // [d][key]  -> key-pairs contiguous (B-frag for PV)
  __shared__ u16 lP[8][16][64];    // wave-private P tiles (bf16)
  int tid  = threadIdx.x;
  int lane = tid & 31, wid = tid >> 5;
  int hf = lane >> 4, mrow = lane & 15;
  int q0 = blockIdx.x * 128;
  int h = blockIdx.y, b = blockIdx.z;
  const u16* Qp = Q + ((size_t)(b*NHEAD + h) * SEQ << 7);
  const u16* Kp = K + ((size_t)(b*NHEAD + h) * SEQ << 7);
  const u16* Vp = V + ((size_t)(b*NHEAD + h) * SEQ << 7);

  // Q fragments (4 x 16x32 over HD=128) loaded straight from global in A-layout
  FragB qf[4];
  int qrow = q0 + wid*16 + mrow;
#pragma unroll
  for (int c = 0; c < 4; ++c)
#pragma unroll
    for (int v = 0; v < 8; ++v) {
      int k = c*32 + ((v >> 2) << 4) + hf*8 + ((v & 3) << 1);
      qf[c].u[v] = *(const u32*)&Qp[((size_t)qrow << 7) + k];
    }

  float mi[8], li[8];
  v8f oacc[8];
#pragma unroll
  for (int r = 0; r < 8; ++r) { mi[r] = -3.0e38f; li[r] = 0.0f; }
#pragma unroll
  for (int t = 0; t < 8; ++t) oacc[t] = {};

  const float sm_scale = 0.088388347648318447f;   // 1/sqrt(128)
  int ntiles = (q0 >> 6) + 2;                     // causal: keys up to q0+127
  for (int jt = 0; jt < ntiles; ++jt) {
    int j0 = jt * 64;
    // K tile 64x128: direct copy -> async-to-LDS
#pragma unroll
    for (int it = 0; it < 4; ++it) {
      int slot = it*256 + tid;
      int row = slot >> 4, cg = slot & 15;
      async_copy_b128(lds_byte_off(&lK[row][cg*8]),
                      &Kp[((size_t)(j0+row) << 7) + cg*8]);
    }
    // V tile 64x128: VGPR-staged transposed store
#pragma unroll
    for (int it = 0; it < 4; ++it) {
      int slot = it*256 + tid;
      int row = slot >> 4, cg = slot & 15;
      uint4 qv = *(const uint4*)&Vp[((size_t)(j0+row) << 7) + cg*8];
      const u16* e = (const u16*)&qv;
#pragma unroll
      for (int jj = 0; jj < 8; ++jj) lVt[cg*8 + jj][row] = e[jj];
    }
    wait_async0();
    __syncthreads();

    // S = Q K^T  (per wave: 16 q-rows x 64 keys = 4 accumulators, K-dim = 128)
    v8f sacc[4];
#pragma unroll
    for (int nt = 0; nt < 4; ++nt) sacc[nt] = {};
#pragma unroll
    for (int c = 0; c < 4; ++c)
#pragma unroll
      for (int nt = 0; nt < 4; ++nt) {
        FragB kb;
#pragma unroll
        for (int v = 0; v < 8; ++v) {
          int dd = c*32 + ((v >> 2) << 4) + hf*8 + ((v & 3) << 1);
          kb.u[v] = *(const u32*)&lK[nt*16 + mrow][dd];
        }
        sacc[nt] = __builtin_amdgcn_wmma_f32_16x16x32_bf16(
            false, qf[c].v, false, kb.v, (short)0, sacc[nt], false, false);
      }

    // scale + causal mask + online softmax (rows live in 16-lane halves)
    float mnew[8];
#pragma unroll
    for (int r = 0; r < 8; ++r) mnew[r] = mi[r];
    int qg_base = q0 + wid*16 + hf*8;
    int kg_base = j0 + mrow;
#pragma unroll
    for (int nt = 0; nt < 4; ++nt)
#pragma unroll
      for (int r = 0; r < 8; ++r) {
        float sv = sacc[nt][r] * sm_scale;
        if (kg_base + nt*16 > qg_base + r) sv = -3.0e38f;
        sacc[nt][r] = sv;
        mnew[r] = fmaxf(mnew[r], sv);
      }
#pragma unroll
    for (int off = 1; off < 16; off <<= 1)
#pragma unroll
      for (int r = 0; r < 8; ++r)
        mnew[r] = fmaxf(mnew[r], __shfl_xor(mnew[r], off, 32));

    float alpha[8], rs[8];
#pragma unroll
    for (int r = 0; r < 8; ++r) { alpha[r] = __expf(mi[r] - mnew[r]); rs[r] = 0.0f; }
#pragma unroll
    for (int nt = 0; nt < 4; ++nt)
#pragma unroll
      for (int r = 0; r < 8; ++r) {
        float p = __expf(sacc[nt][r] - mnew[r]);
        rs[r] += p;
        lP[wid][hf*8 + r][nt*16 + mrow] = f2bf(p);
      }
#pragma unroll
    for (int off = 1; off < 16; off <<= 1)
#pragma unroll
      for (int r = 0; r < 8; ++r)
        rs[r] += __shfl_xor(rs[r], off, 32);
#pragma unroll
    for (int r = 0; r < 8; ++r) { li[r] = li[r]*alpha[r] + rs[r]; mi[r] = mnew[r]; }
#pragma unroll
    for (int t = 0; t < 8; ++t)
#pragma unroll
      for (int r = 0; r < 8; ++r)
        oacc[t][r] *= alpha[r];

    // O += P V   (P reloaded from wave-private LDS in A-layout; same-wave DS is in-order)
#pragma unroll
    for (int c = 0; c < 2; ++c) {
      FragB pa;
#pragma unroll
      for (int v = 0; v < 8; ++v) {
        int k = c*32 + ((v >> 2) << 4) + hf*8 + ((v & 3) << 1);
        pa.u[v] = *(const u32*)&lP[wid][mrow][k];
      }
#pragma unroll
      for (int t = 0; t < 8; ++t) {
        FragB vb;
#pragma unroll
        for (int v = 0; v < 8; ++v) {
          int kk = c*32 + ((v >> 2) << 4) + hf*8 + ((v & 3) << 1);
          vb.u[v] = *(const u32*)&lVt[t*16 + mrow][kk];
        }
        oacc[t] = __builtin_amdgcn_wmma_f32_16x16x32_bf16(
            false, pa.v, false, vb.v, (short)0, oacc[t], false, false);
      }
    }
    __syncthreads();
  }

  // write O flattened [B,S,H] bf16 so the final GEMM reuses the QKV GEMM kernel
#pragma unroll
  for (int r = 0; r < 8; ++r) {
    float invl = 1.0f / li[r];
    int sg = q0 + wid*16 + hf*8 + r;
#pragma unroll
    for (int t = 0; t < 8; ++t) {
      int d = t*16 + mrow;
      O[((size_t)(b*SEQ + sg) << 11) + h*HDIM + d] = f2bf(oacc[t][r] * invl);
    }
  }
}

extern "C" void kernel_launch(void* const* d_in, const int* in_sizes, int n_in,
                              void* d_out, int out_size, void* d_ws, size_t ws_size,
                              hipStream_t stream) {
  const float* hs = (const float*)d_in[0];
  const float* Wq = (const float*)d_in[1];
  const float* Wk = (const float*)d_in[2];
  const float* Wv = (const float*)d_in[3];
  const float* Wo = (const float*)d_in[4];
  float* out = (float*)d_out;

  char* ws = (char*)d_ws;
  size_t off = 0;
  const size_t xbytes = (size_t)MTOT * HID * sizeof(u16);
  const size_t wbytes = (size_t)HID * HID * sizeof(u16);
  u16* Xbf = (u16*)(ws + off); off += xbytes;
  u16* Wqt = (u16*)(ws + off); off += wbytes;
  u16* Wkt = (u16*)(ws + off); off += wbytes;
  u16* Wvt = (u16*)(ws + off); off += wbytes;
  u16* Wot = (u16*)(ws + off); off += wbytes;
  u16* Qb  = (u16*)(ws + off); off += xbytes;
  u16* Kb  = (u16*)(ws + off); off += xbytes;
  u16* Vb  = (u16*)(ws + off); off += xbytes;
  u16* Of  = (u16*)(ws + off); off += xbytes;

  // 1) convert activations, transpose+convert weights
  k_cvt_bf16<<<(MTOT*HID)/256, 256, 0, stream>>>(hs, Xbf, MTOT*HID);
  dim3 tb(32, 8), tg(HID/32, HID/32);
  k_transpose_w<<<tg, tb, 0, stream>>>(Wq, Wqt);
  k_transpose_w<<<tg, tb, 0, stream>>>(Wk, Wkt);
  k_transpose_w<<<tg, tb, 0, stream>>>(Wv, Wvt);
  k_transpose_w<<<tg, tb, 0, stream>>>(Wo, Wot);

  // 2) QKV projections (WMMA), bf16 out in [B,NH,S,HD]
  dim3 gg(HID/128, MTOT/128);
  k_gemm<0><<<gg, 256, 0, stream>>>(Xbf, Wqt, Qb, nullptr);
  k_gemm<0><<<gg, 256, 0, stream>>>(Xbf, Wkt, Kb, nullptr);
  k_gemm<0><<<gg, 256, 0, stream>>>(Xbf, Wvt, Vb, nullptr);

  // 3) RoPE on Q and K
  int ropeN = BATCH * NHEAD * SEQ * 64;
  k_rope<<<ropeN/256, 256, 0, stream>>>(Qb);
  k_rope<<<ropeN/256, 256, 0, stream>>>(Kb);

  // 4) causal flash attention -> O flattened [B,S,H] bf16
  dim3 fg(SEQ/128, NHEAD, BATCH);
  k_flash<<<fg, 256, 0, stream>>>(Qb, Kb, Vb, Of);

  // 5) output projection, f32 straight into d_out
  k_gemm<1><<<gg, 256, 0, stream>>>(Of, Wot, nullptr, out);
}
